// ScalarLongConv_34308198761158
// MI455X (gfx1250) — matrix operand order repaired and checked by reading, not compiled, not run on hardware
//
#include <hip/hip_runtime.h>
#include <hip/hip_bf16.h>

// ---------------------------------------------------------------------------
// Long depthwise conv (K=1025) via Toeplitz-tiled bf16 WMMA on gfx1250.
//   y[b,n,d] = k_mean[b,d] * sum_t q[b, n+t-512, d] * f[d,t]
// A[i,s] = f_d[32w + s - i - 16] is independent of the per-column output
// base, so 16 (b, n-block) streams share one A operand (dense WMMA tiles).
// Staging uses TDM tensor_load_to_lds (A-fragments) and
// global_load_async_to_lds_b128 (q slab); compute is v_wmma_f32_16x16x32_bf16.
// ---------------------------------------------------------------------------

#define B_   8
#define N_   4096
#define D_   512
#define K_   1025
#define NP_  5184          // padded signal length (544 zero pad both sides)
#define NW_  33            // 33 chunks of 32 taps cover K + 15-row skew
#define SLAB_ 1024         // outputs per block (4 tiles of 256 per wave)
#define QROW_ELEMS 2064    // slab halo: [n0-528, n0+1536) per batch row
#define QROW_BYTES 4128    // 2064 * 2
#define QROW_PITCH 4144    // +16B skew: pitch/4 = 1036 ≡ 12 (mod 64) banks
#define SA_BYTES   (NW_ * 1024)          // 33 A-fragments, 1 KB each
#define SQ_OFF     SA_BYTES
#define SMEM_BYTES (SA_BYTES + B_ * QROW_PITCH)   // 66944 B dynamic LDS

typedef __bf16 bf16_t;
typedef __bf16 v16bf __attribute__((ext_vector_type(16)));
typedef float  v8f   __attribute__((ext_vector_type(8)));
typedef unsigned int  u32x4 __attribute__((ext_vector_type(4)));
typedef int           i32x4 __attribute__((ext_vector_type(4)));
typedef int           i32x8 __attribute__((ext_vector_type(8)));

union Frag32B { uint4 u[2]; v16bf v; };

// ---------------------------------------------------------------------------
// 1) k_mean[b,d] = mean_n k[b,n,d]
// ---------------------------------------------------------------------------
__global__ __launch_bounds__(256) void kmean_kernel(const float* __restrict__ k,
                                                    float* __restrict__ km) {
  const int b  = blockIdx.x >> 3;
  const int d0 = (blockIdx.x & 7) * 64;
  const int dl = threadIdx.x & 63;
  const int r  = threadIdx.x >> 6;
  float acc = 0.f;
  for (int n = r; n < N_; n += 4)
    acc += k[((size_t)b * N_ + n) * D_ + d0 + dl];
  __shared__ float red[4][64];
  red[r][dl] = acc;
  __syncthreads();
  if (r == 0)
    km[b * D_ + d0 + dl] =
        (red[0][dl] + red[1][dl] + red[2][dl] + red[3][dl]) * (1.0f / N_);
}

// ---------------------------------------------------------------------------
// 2) zero workspace region (qTp padding must be 0 every call)
// ---------------------------------------------------------------------------
__global__ __launch_bounds__(256) void zero16_kernel(uint4* __restrict__ p, int n16) {
  uint4 z = make_uint4(0u, 0u, 0u, 0u);
  for (int i = blockIdx.x * 256 + threadIdx.x; i < n16; i += gridDim.x * 256)
    p[i] = z;
}

// ---------------------------------------------------------------------------
// 3) pack q (B,N,D f32) -> qTp (D,B,NP bf16), transposed via LDS tiles
// ---------------------------------------------------------------------------
__global__ __launch_bounds__(256) void pack_q_kernel(const float* __restrict__ q,
                                                     bf16_t* __restrict__ qTp) {
  const int blk = blockIdx.x;                 // 8 (d-tiles) * 64 (n-tiles) * 8 (b)
  const int dt = blk & 7;
  const int nt = (blk >> 3) & 63;
  const int b  = blk >> 9;
  const int d0 = dt * 64, n0 = nt * 64;
  __shared__ bf16_t tile[64][72];
  const int jj = threadIdx.x & 63;
  const int r  = threadIdx.x >> 6;
  for (int s = 0; s < 16; ++s) {
    const int i = r + s * 4;
    tile[jj][i] = (bf16_t)q[((size_t)b * N_ + n0 + i) * D_ + d0 + jj];
  }
  __syncthreads();
  const int il = threadIdx.x & 63;
  const int dr = threadIdx.x >> 6;
  for (int s = 0; s < 16; ++s) {
    const int dloc = dr + s * 4;
    qTp[(size_t)((d0 + dloc) * B_ + b) * NP_ + 544 + n0 + il] = tile[dloc][il];
  }
}

// ---------------------------------------------------------------------------
// 4) Toeplitz A-fragments in WMMA register layout:
//    AF[d][w][lane][e] = f_d[32w + s - i - 16],
//    i = lane&15, g = lane>>4, s = e + 16*g   (contiguous k order, matches B)
// ---------------------------------------------------------------------------
__global__ __launch_bounds__(256) void pack_af_kernel(const float* __restrict__ cf,
                                                      bf16_t* __restrict__ AF) {
  const int total = D_ * NW_ * 512;
  for (int idx = blockIdx.x * 256 + threadIdx.x; idx < total; idx += gridDim.x * 256) {
    const int e  = idx & 15;
    const int l  = (idx >> 4) & 31;
    const int wd = idx >> 9;
    const int w  = wd % NW_;
    const int d  = wd / NW_;
    const int i  = l & 15, g = l >> 4;
    const int s  = e + (g << 4);
    const int fi = 32 * w + s - i - 16;
    const float v = (fi >= 0 && fi < K_) ? cf[d * K_ + fi] : 0.0f;
    AF[idx] = (bf16_t)v;
  }
}

// ---------------------------------------------------------------------------
// 5) main WMMA conv: one block = (d, 1024-output slab); 8 waves x 4 tiles.
//    Staging: TDM for A-fragments, async global->LDS for the q slab.
// ---------------------------------------------------------------------------
__global__ __launch_bounds__(256) void conv_wmma_kernel(const bf16_t* __restrict__ qTp,
                                                        const bf16_t* __restrict__ AF,
                                                        float* __restrict__ yT) {
  extern __shared__ __align__(16) unsigned char smem[];
  unsigned char* sA = smem;            // [NW_*1024]
  unsigned char* sQ = smem + SQ_OFF;   // [8][QROW_PITCH]
  const int d  = blockIdx.x >> 2;
  const int n0 = (blockIdx.x & 3) << 10;
  const int tid = threadIdx.x;

  // ---- stage A-fragments for this d (33792 B, one TDM descriptor) ----
  const bf16_t* gA = AF + (size_t)d * (NW_ * 512);
#if __has_builtin(__builtin_amdgcn_tensor_load_to_lds) && __has_builtin(__builtin_amdgcn_s_wait_tensorcnt)
  if (tid == 0) {
    const unsigned long long ga = (unsigned long long)(const void*)gA;
    const unsigned ldsa = (unsigned)(size_t)&sA[0];
    // D# group0: count=1, lds_addr, global_addr[56:0], type=2 (bits 127:126)
    u32x4 g0 = {1u, ldsa, (unsigned)ga,
                ((unsigned)(ga >> 32) & 0x01FFFFFFu) | 0x80000000u};
    // D# group1: data_size=8B (code 3); tensor_dim0=tile_dim0=4224 units;
    // tensor_dim1=tile_dim1=1; tensor_dim0_stride=4224. (bit offsets per ISA §8.4)
    i32x8 g1 = {(int)0x00030000u,            // wg_mask=0, data_size=3
                (int)0x10800000u,            // tensor_dim0[15:0]=0x1080 @bit48
                (int)0x00010000u,            // tensor_dim1=1 @bit80
                (int)0x10800000u,            // tile_dim0=0x1080 @bit112
                (int)0x00000001u,            // tile_dim1=1, tile_dim2=0
                (int)0x00001080u,            // tensor_dim0_stride lo = 4224
                0, 0};
    i32x4 gz4 = {0, 0, 0, 0};
    i32x8 gz8 = {0, 0, 0, 0, 0, 0, 0, 0};
    __builtin_amdgcn_tensor_load_to_lds(g0, g1, gz4, gz4, gz8, 0);
  }
#else
  {
    const uint4* src = (const uint4*)gA;
    uint4* dst = (uint4*)sA;
    for (int u = tid; u < SA_BYTES / 16; u += 256) dst[u] = src[u];
  }
#endif

  // ---- stage q slab: 8 rows x 2064 bf16 = [n0-528, n0+1536) per b ----
  // async global->LDS (ASYNCcnt), 16B per lane, coalesced.
  for (int u = tid; u < B_ * (QROW_BYTES / 16); u += 256) {
    const int b = u / (QROW_BYTES / 16);
    const int o = u - b * (QROW_BYTES / 16);
    const unsigned long long ga =
        (unsigned long long)(const void*)(qTp + (size_t)(d * B_ + b) * NP_ + 16 + n0) +
        (unsigned long long)(o * 16);
    const unsigned ldsa = (unsigned)(size_t)&sQ[b * QROW_PITCH + o * 16];
    asm volatile("global_load_async_to_lds_b128 %0, %1, off"
                 :: "v"(ldsa), "v"(ga) : "memory");
  }
  asm volatile("s_wait_asynccnt 0x0" ::: "memory");
#if __has_builtin(__builtin_amdgcn_tensor_load_to_lds) && __has_builtin(__builtin_amdgcn_s_wait_tensorcnt)
  __builtin_amdgcn_s_wait_tensorcnt(0);
#endif
  __syncthreads();

  // ---- compute: wave j, lane l; 4 tiles per wave share each A fragment ----
  const int l = tid & 31, j = tid >> 5;
  const int i = l & 15, g = l >> 4;
  const int bsig = i >> 1, h = i & 1;          // column -> (batch, half-tile)
  const unsigned aBase = (unsigned)(l * 32);
  const unsigned qBase =
      (unsigned)(bsig * QROW_PITCH + ((32 * j + 16 * h + 16 * g) << 1));

  v8f acc0 = {0.f,0.f,0.f,0.f,0.f,0.f,0.f,0.f};
  v8f acc1 = acc0, acc2 = acc0, acc3 = acc0;
  for (int w = 0; w < NW_; ++w) {
    Frag32B fa;
    const unsigned ab = (unsigned)(w * 1024) + aBase;
    fa.u[0] = *(const uint4*)(sA + ab);
    fa.u[1] = *(const uint4*)(sA + ab + 16);
    const unsigned qb = qBase + (unsigned)(w * 64);
    Frag32B f0, f1, f2, f3;                    // tiles at +0,+256,+512,+768 outputs
    f0.u[0] = *(const uint4*)(sQ + qb);        f0.u[1] = *(const uint4*)(sQ + qb + 16);
    f1.u[0] = *(const uint4*)(sQ + qb + 512);  f1.u[1] = *(const uint4*)(sQ + qb + 528);
    f2.u[0] = *(const uint4*)(sQ + qb + 1024); f2.u[1] = *(const uint4*)(sQ + qb + 1040);
    f3.u[0] = *(const uint4*)(sQ + qb + 1536); f3.u[1] = *(const uint4*)(sQ + qb + 1552);
    acc0 = __builtin_amdgcn_wmma_f32_16x16x32_bf16(false, fa.v, false, f0.v, (short)0, acc0, false, false);
    acc1 = __builtin_amdgcn_wmma_f32_16x16x32_bf16(false, fa.v, false, f1.v, (short)0, acc1, false, false);
    acc2 = __builtin_amdgcn_wmma_f32_16x16x32_bf16(false, fa.v, false, f2.v, (short)0, acc2, false, false);
    acc3 = __builtin_amdgcn_wmma_f32_16x16x32_bf16(false, fa.v, false, f3.v, (short)0, acc3, false, false);
  }

  // D layout: VGPR r -> row M = r + 8g; column N = lane&15.
  float* row = yT + (size_t)(d * B_ + bsig) * N_ + (n0 + 32 * j + 16 * h + 8 * g);
  v8f* accs[4] = {&acc0, &acc1, &acc2, &acc3};
  for (int t = 0; t < 4; ++t) {
    v8f a = *accs[t];
    float* dst = row + 256 * t;
    *(float4*)(dst)     = make_float4(a[0], a[1], a[2], a[3]);
    *(float4*)(dst + 4) = make_float4(a[4], a[5], a[6], a[7]);
  }
}

// ---------------------------------------------------------------------------
// 6) yT (D,B,N) -> out (B,N,D), scaled by k_mean, coalesced both sides
// ---------------------------------------------------------------------------
__global__ __launch_bounds__(256) void out_kernel(const float* __restrict__ yT,
                                                  const float* __restrict__ km,
                                                  float* __restrict__ out) {
  const int blk = blockIdx.x;                 // 16 (d) * 128 (n) * 8 (b)
  const int dt = blk & 15;
  const int nt = (blk >> 4) & 127;
  const int b  = blk >> 11;
  const int d0 = dt * 32, n0 = nt * 32;
  __shared__ float tile[32][33];
  const int nl = threadIdx.x & 31;
  const int r  = threadIdx.x >> 5;
  for (int s = 0; s < 4; ++s) {
    const int dl = r + s * 8;
    tile[dl][nl] = yT[((size_t)(d0 + dl) * B_ + b) * N_ + n0 + nl];
  }
  __syncthreads();
  const int dl = threadIdx.x & 31;
  const int nr = threadIdx.x >> 5;
  const float sc = km[b * D_ + d0 + dl];
  for (int s = 0; s < 4; ++s) {
    const int n = n0 + nr + s * 8;
    out[((size_t)b * N_ + n) * D_ + d0 + dl] = sc * tile[dl][nr + s * 8];
  }
}

// ---------------------------------------------------------------------------
extern "C" void kernel_launch(void* const* d_in, const int* in_sizes, int n_in,
                              void* d_out, int out_size, void* d_ws, size_t ws_size,
                              hipStream_t stream) {
  const float* q  = (const float*)d_in[0];
  const float* k  = (const float*)d_in[1];
  const float* cf = (const float*)d_in[2];
  float* out = (float*)d_out;

  unsigned char* ws = (unsigned char*)d_ws;
  const size_t qtp_bytes = (size_t)D_ * B_ * NP_ * 2;   // 42.5 MB
  const size_t af_bytes  = (size_t)D_ * NW_ * 512 * 2;  // 17.3 MB
  float*  kmean = (float*)ws;
  bf16_t* qTp   = (bf16_t*)(ws + 16384);
  bf16_t* AF    = (bf16_t*)(ws + 16384 + qtp_bytes);
  float*  yT    = (float*)(ws + 16384 + qtp_bytes + af_bytes);   // 67 MB
  (void)in_sizes; (void)n_in; (void)out_size; (void)ws_size;

  kmean_kernel<<<dim3(B_ * (D_ / 64)), dim3(256), 0, stream>>>(k, kmean);
  zero16_kernel<<<dim3(2048), dim3(256), 0, stream>>>((uint4*)qTp, (int)(qtp_bytes / 16));
  pack_q_kernel<<<dim3(8 * 64 * 8), dim3(256), 0, stream>>>(q, qTp);
  pack_af_kernel<<<dim3(4096), dim3(256), 0, stream>>>(cf, AF);
  conv_wmma_kernel<<<dim3(D_ * (N_ / SLAB_)), dim3(256), SMEM_BYTES, stream>>>(qTp, AF, yT);
  out_kernel<<<dim3(16 * 128 * 8), dim3(256), 0, stream>>>(yT, kmean, out);
}
